// STGCNBlock_33423435497812
// MI455X (gfx1250) — compile-verified, matrix operand order checked by reading, not run on hardware
//
#include <hip/hip_runtime.h>
#include <hip/hip_bf16.h>

typedef __attribute__((ext_vector_type(16))) _Float16 v16h;
typedef __attribute__((ext_vector_type(8)))  float    v8f;
typedef __attribute__((ext_vector_type(4)))  unsigned int v4u;
typedef __attribute__((ext_vector_type(8)))  int      v8i;
typedef __attribute__((ext_vector_type(4)))  int      v4i;

#define Bdim 32
#define Tdim 256
#define Ndim 25
#define CIN  64
#define COUT 128
// M = B*T*N = 204800 rows, 12800 tiles of 16

// ---------------- WMMA fragment helpers (CDNA5 ISA 7.12.2 layouts) ----------
// Lane L: l = L%16, h = L/16.
// A (16x32, MxK): e<8 -> K=8h+e ; e>=8 -> K=16+8h+(e-8); M=l.
// B (32x16, KxN): same K pattern per lane, lane indexes column N=l.

__device__ __forceinline__ v16h load_frag_f16(const _Float16* src, int row0, int ld,
                                              int k0, int l, int h) {
  v16h a;
  const _Float16* p = src + (size_t)(row0 + l) * ld + k0 + 8 * h;
#pragma unroll
  for (int e = 0; e < 8; ++e) a[e] = p[e];
#pragma unroll
  for (int e = 0; e < 8; ++e) a[8 + e] = p[16 + e];
  return a;
}

__device__ __forceinline__ v16h load_frag_f32(const float* src, int row0, int ld,
                                              int k0, int l, int h) {
  v16h a;
  const float* p = src + (size_t)(row0 + l) * ld + k0 + 8 * h;
#pragma unroll
  for (int e = 0; e < 8; ++e) a[e] = (_Float16)p[e];
#pragma unroll
  for (int e = 0; e < 8; ++e) a[8 + e] = (_Float16)p[16 + e];
  return a;
}

__device__ __forceinline__ v8f wmma_f16(v16h a, v16h b, v8f c) {
  return __builtin_amdgcn_wmma_f32_16x16x32_f16(false, a, false, b, (short)0, c,
                                                false, false);
}

// ------------- Tensor Data Mover: 2D f16 tile (nrows x rowlen) -> LDS -------
// D# per CDNA5 ISA ch.8: group0 {count=1, lds_addr, global_addr[56:0], type=2},
// group1 {data_size=2B, tensor_dim0=rowlen, tensor_dim1=nrows, tile_dim0=rowlen,
//         tile_dim1=nrows, tensor_dim0_stride=rowlen}. Groups 2/3 unused (2D),
// passed as zeros; this toolchain uses the 6-arg builtin (extra int32x8 chunk).
__device__ __forceinline__ void tdm_load_2d_f16(unsigned lds_off,
                                                const _Float16* gptr,
                                                int nrows, int rowlen) {
  unsigned long long ga = (unsigned long long)gptr;
  v4u g0;
  g0[0] = 1u;                                             // count=1, user desc
  g0[1] = lds_off;                                        // LDS byte address
  g0[2] = (unsigned)(ga & 0xFFFFFFFFull);                 // global_addr[31:0]
  g0[3] = (unsigned)((ga >> 32) & 0x01FFFFFFull) | (2u << 30);  // [56:32]|type=2
  v8i g1;
  g1[0] = (int)(1u << 16);                                // data_size=1 (2 bytes)
  g1[1] = (int)(((unsigned)rowlen & 0xFFFFu) << 16);      // tensor_dim0[15:0]
  g1[2] = (int)(((unsigned)nrows  & 0xFFFFu) << 16);      // dim0 hi=0 | tensor_dim1
  g1[3] = (int)(((unsigned)rowlen & 0xFFFFu) << 16);      // dim1 hi=0 | tile_dim0
  g1[4] = (int)((unsigned)nrows & 0xFFFFu);               // tile_dim1 | tile_dim2=0
  g1[5] = rowlen;                                         // tensor_dim0_stride lo
  g1[6] = 0;
  g1[7] = 0;
  v4i gz4 = {0, 0, 0, 0};
  v8i gz8 = {0, 0, 0, 0, 0, 0, 0, 0};
  __builtin_amdgcn_tensor_load_to_lds(g0, g1, gz4, gz4, gz8, 0);
}

// ---- K0: a = adj * softmax(att, axis=1) (25x25) ----
__global__ void k_build_a(const float* __restrict__ adj,
                          const float* __restrict__ att,
                          float* __restrict__ a_out) {
  int i = threadIdx.x;
  if (i < Ndim) {
    float row[Ndim];
    float mx = -1e30f;
    for (int j = 0; j < Ndim; ++j) { row[j] = att[i * Ndim + j]; mx = fmaxf(mx, row[j]); }
    float s = 0.f;
    for (int j = 0; j < Ndim; ++j) { row[j] = __expf(row[j] - mx); s += row[j]; }
    float inv = 1.f / s;
    for (int j = 0; j < Ndim; ++j)
      a_out[i * Ndim + j] = adj[i * Ndim + j] * row[j] * inv;
  }
}

// ---- K0b: one-time f32 -> f16 weight conversion (+ tconv de-striding) ----
// Layout in wbuf (halves): conv[128x64] @0, tconv[tap][128x128] @8192,
// qkv[384x128] @57344, proj[128x128] @106496, res[128x64] @122880.
__global__ void k_prep(const float* __restrict__ conv_w,
                       const float* __restrict__ tconv_w,
                       const float* __restrict__ qkv_w,
                       const float* __restrict__ proj_w,
                       const float* __restrict__ res_w,
                       _Float16* __restrict__ wbuf) {
  for (int i = blockIdx.x * 256 + threadIdx.x; i < 131072; i += gridDim.x * 256) {
    float v;
    if (i < 8192) {
      v = conv_w[i];
    } else if (i < 57344) {
      int r = i - 8192;
      int tap = r / 16384, rem = r % 16384;     // rem = o*128 + ci
      v = tconv_w[rem * 3 + tap];
    } else if (i < 106496) {
      v = qkv_w[i - 57344];
    } else if (i < 122880) {
      v = proj_w[i - 106496];
    } else {
      v = res_w[i - 122880];
    }
    wbuf[i] = (_Float16)v;
  }
}

// ---- K1: spatial mix + transpose to (B,N,T,CIN) f16 ----
__global__ void k_mix(const float* __restrict__ x, const float* __restrict__ a,
                      _Float16* __restrict__ xmix) {
  __shared__ float xs_s[Ndim * CIN];
  __shared__ float a_s[Ndim * Ndim];
  int bt = blockIdx.x;            // b*T + t
  int b = bt / Tdim, t = bt % Tdim;
  const float* xp = x + (size_t)bt * (Ndim * CIN);
  for (int i = threadIdx.x; i < Ndim * CIN; i += 256) xs_s[i] = xp[i];
  for (int i = threadIdx.x; i < Ndim * Ndim; i += 256) a_s[i] = a[i];
  __syncthreads();
  for (int i = threadIdx.x; i < Ndim * CIN; i += 256) {
    int m = i >> 6, c = i & 63;
    float acc = 0.f;
#pragma unroll
    for (int n = 0; n < Ndim; ++n) acc += xs_s[n * CIN + c] * a_s[n * Ndim + m];
    xmix[(((size_t)b * Ndim + m) * Tdim + t) * CIN + c] = (_Float16)acc;
  }
}

// ---- K2: xs = relu(xmix @ conv_w^T + conv_b)  [M,64] x [64,128] ----
__global__ void k_conv1x1(const _Float16* __restrict__ xmix,
                          const _Float16* __restrict__ Wh,
                          const float* __restrict__ bias,
                          _Float16* __restrict__ xs) {
  int mt = blockIdx.x;
  int wave = threadIdx.x >> 5, lane = threadIdx.x & 31;
  int l = lane & 15, h = lane >> 4;
  int m0 = mt * 16, n0 = wave * 16;
  __builtin_prefetch(xmix + (size_t)(m0 + 16) * CIN, 0, 1);
  v8f c = {};
#pragma unroll
  for (int k0 = 0; k0 < CIN; k0 += 32) {
    v16h a  = load_frag_f16(xmix, m0, CIN, k0, l, h);
    v16h bf = load_frag_f16(Wh, n0, CIN, k0, l, h);
    c = wmma_f16(a, bf, c);
  }
  float bsv = bias[n0 + l];
#pragma unroll
  for (int r = 0; r < 8; ++r) {
    float v = fmaxf(c[r] + bsv, 0.f);
    xs[(size_t)(m0 + r + 8 * h) * COUT + n0 + l] = (_Float16)v;
  }
}

// ---- K3: temporal conv K=3 along t within each (b,n), + relu ----
// A-tile (18 halo rows x 128) staged to LDS once per block via the TDM.
__global__ void k_tconv(const _Float16* __restrict__ xs,
                        const _Float16* __restrict__ Wt,   // [tap][128x128] f16
                        const float* __restrict__ bias,
                        _Float16* __restrict__ xt) {
  __shared__ _Float16 tile[18 * COUT];                     // rows t0-1 .. t0+16
  int mt = blockIdx.x;
  int bn = mt >> 4;
  int t0 = (mt & 15) * 16;
  int wave = threadIdx.x >> 5, lane = threadIdx.x & 31;
  int l = lane & 15, h = lane >> 4;
  int n0 = wave * 16;

  // zero halo rows that fall outside [0,T)
  if (t0 == 0) {
    for (int i = threadIdx.x; i < COUT; i += 256) tile[i] = (_Float16)0.f;
  }
  if (t0 == Tdim - 16) {
    for (int i = threadIdx.x; i < COUT; i += 256) tile[17 * COUT + i] = (_Float16)0.f;
  }
  if (threadIdx.x < 32) {                                  // wave 0 drives the TDM
    int rs = (t0 == 0) ? 0 : t0 - 1;
    int re = (t0 == Tdim - 16) ? Tdim - 1 : t0 + 16;
    int nrows = re - rs + 1;                               // 17 or 18
    int dst_row = rs - (t0 - 1);                           // 0 or 1
    unsigned lds_off = (unsigned)(unsigned long long)&tile[dst_row * COUT];
    const _Float16* gsrc = xs + ((size_t)bn * Tdim + rs) * COUT;
    tdm_load_2d_f16(lds_off, gsrc, nrows, COUT);
    __builtin_amdgcn_s_wait_tensorcnt(0);
  }
  __syncthreads();

  v8f c = {};
#pragma unroll
  for (int tap = 0; tap < 3; ++tap) {
    const _Float16* wtap = Wt + tap * (COUT * COUT);
#pragma unroll
    for (int k0 = 0; k0 < COUT; k0 += 32) {
      // LDS row (l + tap) holds t = t0 + l + tap - 1
      v16h a  = load_frag_f16(tile, tap, COUT, k0, l, h);
      v16h bf = load_frag_f16(wtap, n0, COUT, k0, l, h);
      c = wmma_f16(a, bf, c);
    }
  }
  float bsv = bias[n0 + l];
#pragma unroll
  for (int r = 0; r < 8; ++r) {
    float v = fmaxf(c[r] + bsv, 0.f);
    xt[((size_t)bn * Tdim + t0 + r + 8 * h) * COUT + n0 + l] = (_Float16)v;
  }
}

// ---- K4: fused qkv GEMM (K=128, N=384) + per-token 4x4 head attention ----
__global__ void k_attn(const _Float16* __restrict__ xt,
                       const _Float16* __restrict__ Wq,    // [384x128] f16
                       const float* __restrict__ bqkv,
                       _Float16* __restrict__ o) {
  __shared__ float qkv_s[16][384];
  __shared__ _Float16 atile[16 * COUT];
  int mt = blockIdx.x;
  int m0 = mt * 16;
  int wave = threadIdx.x >> 5, lane = threadIdx.x & 31;
  int l = lane & 15, h = lane >> 4;

  if (threadIdx.x < 32) {                                  // TDM stage A tile
    unsigned lds_off = (unsigned)(unsigned long long)&atile[0];
    tdm_load_2d_f16(lds_off, xt + (size_t)m0 * COUT, 16, COUT);
    __builtin_amdgcn_s_wait_tensorcnt(0);
  }
  __syncthreads();

  v16h afr[4];
#pragma unroll
  for (int ki = 0; ki < 4; ++ki) afr[ki] = load_frag_f16(atile, 0, COUT, ki * 32, l, h);
#pragma unroll
  for (int s = 0; s < 3; ++s) {
    int n0 = (wave * 3 + s) * 16;
    v8f c = {};
#pragma unroll
    for (int ki = 0; ki < 4; ++ki) {
      v16h bf = load_frag_f16(Wq, n0, COUT, ki * 32, l, h);
      c = wmma_f16(afr[ki], bf, c);
    }
    float bb = bqkv[n0 + l];
#pragma unroll
    for (int r = 0; r < 8; ++r) qkv_s[r + 8 * h][n0 + l] = c[r] + bb;
  }
  __syncthreads();
  int tid = threadIdx.x;
  if (tid < 64) {                       // one thread per (row, head)
    int r = tid >> 2, hh = tid & 3;
    const float* q  = &qkv_s[r][hh * 32];
    const float* kk = &qkv_s[r][128];
    const float* vv = &qkv_s[r][256];
    float sc[4];
    float mx = -1e30f;
#pragma unroll
    for (int j = 0; j < 4; ++j) {
      float d = 0.f;
#pragma unroll
      for (int e = 0; e < 32; ++e) d += q[e] * kk[j * 32 + e];
      sc[j] = d * 0.17677669529663687f;   // 1/sqrt(32)
      mx = fmaxf(mx, sc[j]);
    }
    float p[4], ssum = 0.f;
#pragma unroll
    for (int j = 0; j < 4; ++j) { p[j] = __expf(sc[j] - mx); ssum += p[j]; }
    float inv = 1.f / ssum;
#pragma unroll
    for (int e = 0; e < 32; ++e) {
      float acc = 0.f;
#pragma unroll
      for (int j = 0; j < 4; ++j) acc += p[j] * vv[j * 32 + e];
      o[(size_t)(m0 + r) * COUT + hh * 32 + e] = (_Float16)(acc * inv);
    }
  }
}

// ---- K5: out = o @ proj^T + proj_b + x @ res_w^T + res_b (f32 out) ----
// Linear row l of bnt-layout `o` pairs with linear row l of btn-layout x/out:
// the reference's swapaxes+reshape is the identity on flat row indices.
__global__ void k_final(const _Float16* __restrict__ o, const float* __restrict__ x,
                        const _Float16* __restrict__ Wp, const float* __restrict__ bp,
                        const _Float16* __restrict__ Wr, const float* __restrict__ br,
                        float* __restrict__ out) {
  int mt = blockIdx.x;
  int wave = threadIdx.x >> 5, lane = threadIdx.x & 31;
  int l = lane & 15, h = lane >> 4;
  int m0 = mt * 16, n0 = wave * 16;
  __builtin_prefetch(o + (size_t)(m0 + 16) * COUT, 0, 1);
  v8f c = {};
#pragma unroll
  for (int k0 = 0; k0 < COUT; k0 += 32) {   // projection path
    v16h a  = load_frag_f16(o, m0, COUT, k0, l, h);
    v16h bf = load_frag_f16(Wp, n0, COUT, k0, l, h);
    c = wmma_f16(a, bf, c);
  }
#pragma unroll
  for (int k0 = 0; k0 < CIN; k0 += 32) {    // residual path
    v16h a  = load_frag_f32(x, m0, CIN, k0, l, h);
    v16h bf = load_frag_f16(Wr, n0, CIN, k0, l, h);
    c = wmma_f16(a, bf, c);
  }
  float bb = bp[n0 + l] + br[n0 + l];
#pragma unroll
  for (int r = 0; r < 8; ++r)
    out[(size_t)(m0 + r + 8 * h) * COUT + n0 + l] = c[r] + bb;
}

extern "C" void kernel_launch(void* const* d_in, const int* in_sizes, int n_in,
                              void* d_out, int out_size, void* d_ws, size_t ws_size,
                              hipStream_t stream) {
  const float* x       = (const float*)d_in[0];
  const float* adj     = (const float*)d_in[1];
  const float* att     = (const float*)d_in[2];
  const float* conv_w  = (const float*)d_in[3];
  const float* conv_b  = (const float*)d_in[4];
  const float* tconv_w = (const float*)d_in[5];
  const float* tconv_b = (const float*)d_in[6];
  const float* qkv_w   = (const float*)d_in[7];
  const float* qkv_b   = (const float*)d_in[8];
  const float* proj_w  = (const float*)d_in[9];
  const float* proj_b  = (const float*)d_in[10];
  const float* res_w   = (const float*)d_in[11];
  const float* res_b   = (const float*)d_in[12];

  const size_t M   = (size_t)Bdim * Tdim * Ndim;        // 204800
  const size_t BUF = M * COUT * sizeof(_Float16);       // 52,428,800 B
  char* ws = (char*)d_ws;
  float*    a_mat = (float*)ws;                         // 2.5 KB @ 0
  _Float16* wbuf  = (_Float16*)(ws + 4096);             // 262,144 B of f16 weights
  _Float16* r1    = (_Float16*)(ws + 4096 + 262144);    // xmix -> xt
  _Float16* r2    = (_Float16*)(ws + 4096 + 262144 + BUF);  // xs -> o

  _Float16* w_conv = wbuf;                              // 128x64
  _Float16* w_t    = wbuf + 8192;                       // 3 x 128x128
  _Float16* w_q    = wbuf + 57344;                      // 384x128
  _Float16* w_p    = wbuf + 106496;                     // 128x128
  _Float16* w_r    = wbuf + 122880;                     // 128x64

  const int MT = (int)(M / 16);                         // 12800 row tiles

  k_build_a<<<1, 32, 0, stream>>>(adj, att, a_mat);
  k_prep   <<<128, 256, 0, stream>>>(conv_w, tconv_w, qkv_w, proj_w, res_w, wbuf);
  k_mix    <<<Bdim * Tdim, 256, 0, stream>>>(x, a_mat, r1);
  k_conv1x1<<<MT, 256, 0, stream>>>(r1, w_conv, conv_b, r2);
  k_tconv  <<<MT, 256, 0, stream>>>(r2, w_t, tconv_b, r1);
  k_attn   <<<MT, 256, 0, stream>>>(r1, w_q, qkv_b, r2);
  k_final  <<<MT, 256, 0, stream>>>(r2, x, w_p, proj_b, w_r, res_b,
                                    (float*)d_out);
}
